// KdModel_59957743452328
// MI455X (gfx1250) — matrix operand order repaired and checked by reading, not compiled, not run on hardware
//
#include <hip/hip_runtime.h>
#include <hip/hip_bf16.h>

#define NN 50000      // nodes
#define NE 800000     // edges
#define DD 64         // feature dim
#define NL 3          // layers
#define NG 32         // graphs
#define EPS_BN 1e-5f
#define NEG_SLOPE 0.2f

typedef __attribute__((ext_vector_type(16))) _Float16 v16h;
typedef __attribute__((ext_vector_type(8)))  _Float16 v8h;
typedef __attribute__((ext_vector_type(4)))  _Float16 v4h;
typedef __attribute__((ext_vector_type(8)))  float    v8f;

// ---------------------------------------------------------------------------
// WMMA fragment loaders (CDNA5 16x16x32 f16 layouts, ISA 7.12.2), vectorized:
// A (row-major in LDS): lane m=lane%16, kh=lane/16; halfwords {kh*8..+7, 16+kh*8..+7}
//   -> two contiguous 16B chunks => 2x ds_load_b128
// B staged TRANSPOSED (n-major) in LDS: lane n=lane%16, halfwords {kh*8..+7, 16+kh*8..+7}
//   -> two contiguous 16B chunks => 2x ds_load_b128
// C/D: vgpr v -> M = v + 8*kh, col N = lane%16
// ---------------------------------------------------------------------------
__device__ __forceinline__ v16h frag_a(const _Float16* base, int ld, int lane) {
  const int m = lane & 15, kh = (lane >> 4) & 1;
  const _Float16* p = base + m * ld + kh * 8;
  v8h lo = *(const v8h*)p;
  v8h hi = *(const v8h*)(p + 16);
  return __builtin_shufflevector(lo, hi, 0,1,2,3,4,5,6,7,8,9,10,11,12,13,14,15);
}
__device__ __forceinline__ v16h frag_bt(const _Float16* baseT, int ldk, int lane) {
  const int n = lane & 15, kh = (lane >> 4) & 1;
  const _Float16* p = baseT + n * ldk + kh * 8;
  v8h lo = *(const v8h*)p;
  v8h hi = *(const v8h*)(p + 16);
  return __builtin_shufflevector(lo, hi, 0,1,2,3,4,5,6,7,8,9,10,11,12,13,14,15);
}

// order-preserving float<->uint encoding for atomic segment-max
__device__ __forceinline__ unsigned encF(float f) {
  unsigned u = __float_as_uint(f);
  return (u & 0x80000000u) ? ~u : (u | 0x80000000u);
}
__device__ __forceinline__ float decF(unsigned u) {
  return __uint_as_float((u & 0x80000000u) ? (u & 0x7FFFFFFFu) : ~u);
}

// ---------------------------------------------------------------------------
// Edge MLP: out = relu(cat(x[row],x[col],e_in) @ W1 + b1) @ W2 + b2
// 256 thr (8 waves); weights resident in LDS (transposed); grid-stride over
// groups of 32 edges. Wave w -> rowtile w>>2 (16 edges), coltile w&3 (16 cols).
// ---------------------------------------------------------------------------
__global__ __launch_bounds__(256) void k_edge_mlp(
    const float* __restrict__ x, const int* __restrict__ row,
    const int* __restrict__ col, const float* __restrict__ e_in,
    const float* __restrict__ w1, const float* __restrict__ b1,
    const float* __restrict__ w2, const float* __restrict__ b2,
    float* __restrict__ e_out) {
  __shared__ __align__(16) _Float16 sW1t[64 * 192];  // transposed: [n][k], 24 KB
  __shared__ __align__(16) _Float16 sW2t[64 * 64];   // transposed: [n][k],  8 KB
  __shared__ __align__(16) _Float16 sA[32 * 192];    // 12 KB
  __shared__ __align__(16) _Float16 sEH[32 * 64];    //  4 KB
  __shared__ int sRow[32], sCol[32];

  const int tid  = threadIdx.x;
  const int lane = tid & 31;
  const int wv   = tid >> 5;
  const int rt   = wv >> 2;       // 0..1
  const int ct   = wv & 3;        // 0..3
  const int n    = ct * 16 + (lane & 15);
  const int kh   = (lane >> 4) & 1;

  // stage weights transposed as f16 (once per block; contiguous v4h LDS stores)
  for (int j = tid; j < 64 * 48; j += 256) {     // W1: n = j/48, k = 4*(j%48)
    const int nn = j / 48, k = (j - nn * 48) * 4;
    v4h t = { (_Float16)w1[(k + 0) * 64 + nn], (_Float16)w1[(k + 1) * 64 + nn],
              (_Float16)w1[(k + 2) * 64 + nn], (_Float16)w1[(k + 3) * 64 + nn] };
    *(v4h*)(sW1t + nn * 192 + k) = t;
  }
  for (int j = tid; j < 64 * 16; j += 256) {     // W2: n = j/16, k = 4*(j%16)
    const int nn = j >> 4, k = (j & 15) * 4;
    v4h t = { (_Float16)w2[(k + 0) * 64 + nn], (_Float16)w2[(k + 1) * 64 + nn],
              (_Float16)w2[(k + 2) * 64 + nn], (_Float16)w2[(k + 3) * 64 + nn] };
    *(v4h*)(sW2t + nn * 64 + k) = t;
  }
  const float bia1 = b1[n];
  const float bia2 = b2[n];

  for (int g = blockIdx.x; g < NE / 32; g += gridDim.x) {
    const int e0 = g * 32;
    __syncthreads();   // covers weight staging (1st iter) + sA/sEH reuse

    if (tid < 32) {
      sRow[tid] = row[e0 + tid];
      sCol[tid] = col[e0 + tid];
    }
    __syncthreads();

    // gather cat(x[row], x[col], e_attr) -> sA (32 x 192), float4 -> v4h
    for (int c = tid; c < 32 * 48; c += 256) {
      const int e = c / 48, q = c - e * 48;      // q = float4 index within row
      float4 v;
      if (q < 16)       v = ((const float4*)(x + (size_t)sRow[e] * DD))[q];
      else if (q < 32)  v = ((const float4*)(x + (size_t)sCol[e] * DD))[q - 16];
      else              v = ((const float4*)(e_in + (size_t)(e0 + e) * DD))[q - 32];
      v4h t = { (_Float16)v.x, (_Float16)v.y, (_Float16)v.z, (_Float16)v.w };
      *(v4h*)(sA + e * 192 + q * 4) = t;
    }
    __syncthreads();

    // GEMM1: (16x192)@(192x16) as 6 WMMA over K=32
    v8f acc = {};
#pragma unroll
    for (int kk = 0; kk < 6; ++kk) {
      v16h a = frag_a(sA + rt * 16 * 192 + kk * 32, 192, lane);
      v16h b = frag_bt(sW1t + (ct * 16) * 192 + kk * 32, 192, lane);
      acc = __builtin_amdgcn_wmma_f32_16x16x32_f16(false, a, false, b,
                                                   (short)0, acc, false, false);
    }
    // bias + relu -> sEH (f16)
#pragma unroll
    for (int v = 0; v < 8; ++v) {
      float t = acc[v] + bia1;
      t = t > 0.f ? t : 0.f;
      sEH[(rt * 16 + v + 8 * kh) * 64 + n] = (_Float16)t;
    }
    __syncthreads();

    // GEMM2: (16x64)@(64x16) as 2 WMMA
    v8f acc2 = {};
#pragma unroll
    for (int kk = 0; kk < 2; ++kk) {
      v16h a = frag_a(sEH + rt * 16 * 64 + kk * 32, 64, lane);
      v16h b = frag_bt(sW2t + (ct * 16) * 64 + kk * 32, 64, lane);
      acc2 = __builtin_amdgcn_wmma_f32_16x16x32_f16(false, a, false, b,
                                                    (short)0, acc2, false, false);
    }
#pragma unroll
    for (int v = 0; v < 8; ++v) {
      const int e = e0 + rt * 16 + v + 8 * kh;
      e_out[(size_t)e * DD + n] = acc2[v] + bia2;
    }
  }
}

// ---------------------------------------------------------------------------
// xw = x @ gat_w  (N x 64) @ (64 x 64): weights resident, grid-stride groups of 32
// ---------------------------------------------------------------------------
__global__ __launch_bounds__(256) void k_xw(
    const float* __restrict__ x, const float* __restrict__ w,
    float* __restrict__ xw) {
  __shared__ __align__(16) _Float16 sWt[64 * 64];   // transposed [n][k]
  __shared__ __align__(16) _Float16 sA[32 * 64];
  const int tid = threadIdx.x, lane = tid & 31, wv = tid >> 5;
  const int rt = wv >> 2, ct = wv & 3;
  const int n = ct * 16 + (lane & 15);
  const int kh = (lane >> 4) & 1;
  const int ngroups = (NN + 31) / 32;

  for (int j = tid; j < 64 * 16; j += 256) {
    const int nn = j >> 4, k = (j & 15) * 4;
    v4h t = { (_Float16)w[(k + 0) * 64 + nn], (_Float16)w[(k + 1) * 64 + nn],
              (_Float16)w[(k + 2) * 64 + nn], (_Float16)w[(k + 3) * 64 + nn] };
    *(v4h*)(sWt + nn * 64 + k) = t;
  }

  for (int g = blockIdx.x; g < ngroups; g += gridDim.x) {
    const int i0 = g * 32;
    __syncthreads();
    for (int c = tid; c < 32 * 16; c += 256) {
      const int e = c >> 4, q = c & 15;
      int gi = i0 + e; if (gi >= NN) gi = NN - 1;
      float4 v = ((const float4*)(x + (size_t)gi * DD))[q];
      v4h t = { (_Float16)v.x, (_Float16)v.y, (_Float16)v.z, (_Float16)v.w };
      *(v4h*)(sA + e * 64 + q * 4) = t;
    }
    __syncthreads();

    v8f acc = {};
#pragma unroll
    for (int kk = 0; kk < 2; ++kk) {
      v16h a = frag_a(sA + rt * 16 * 64 + kk * 32, 64, lane);
      v16h b = frag_bt(sWt + (ct * 16) * 64 + kk * 32, 64, lane);
      acc = __builtin_amdgcn_wmma_f32_16x16x32_f16(false, a, false, b,
                                                   (short)0, acc, false, false);
    }
#pragma unroll
    for (int v = 0; v < 8; ++v) {
      const int i = i0 + rt * 16 + v + 8 * kh;
      if (i < NN) xw[(size_t)i * DD + n] = acc[v];
    }
  }
}

// per-node attention scalars: asrc = xw . att_src, adst = xw . att_dst
__global__ void k_node_attn(const float* __restrict__ xw,
                            const float* __restrict__ as,
                            const float* __restrict__ ad,
                            float* __restrict__ asrc,
                            float* __restrict__ adst) {
  const int i = blockIdx.x * blockDim.x + threadIdx.x;
  if (i >= NN) return;
  const float4* r = (const float4*)(xw + (size_t)i * DD);
  const float4* a1 = (const float4*)as;
  const float4* a2 = (const float4*)ad;
  float s = 0.f, d = 0.f;
#pragma unroll
  for (int k = 0; k < 16; ++k) {
    const float4 v = r[k], p = a1[k], q = a2[k];
    s += v.x * p.x + v.y * p.y + v.z * p.z + v.w * p.w;
    d += v.x * q.x + v.y * q.y + v.z * q.z + v.w * q.w;
  }
  asrc[i] = s; adst[i] = d;
}

// v = edge_w @ att_edge  (64)
__global__ void k_edge_vec(const float* __restrict__ ew,
                           const float* __restrict__ ae,
                           float* __restrict__ v) {
  const int i = threadIdx.x;
  if (i >= DD) return;
  float s = 0.f;
#pragma unroll 8
  for (int j = 0; j < DD; ++j) s += ew[i * DD + j] * ae[j];
  v[i] = s;
}

// init per-layer node state: h = bias (broadcast), s = 0, m_enc = 0
__global__ void k_init_node(float* __restrict__ s, unsigned* __restrict__ menc,
                            float* __restrict__ h, const float* __restrict__ bias) {
  const size_t idx = (size_t)blockIdx.x * blockDim.x + threadIdx.x;
  if (idx >= (size_t)NN * DD) return;
  const int d = (int)(idx & 63);
  h[idx] = bias[d];
  if (d == 0) {
    const size_t i = idx >> 6;
    s[i] = 0.f;
    menc[i] = 0u;   // decodes to -NaN -> "no edge", matches -inf semantics
  }
}

// logits + leaky relu + segment max (encoded atomicMax)
__global__ void k_edge_logits(const float* __restrict__ ea,
                              const float* __restrict__ v,
                              const float* __restrict__ asrc,
                              const float* __restrict__ adst,
                              const int* __restrict__ row,
                              const int* __restrict__ col,
                              float* __restrict__ logit,
                              unsigned* __restrict__ menc) {
  const int e = blockIdx.x * blockDim.x + threadIdx.x;
  if (e >= NE) return;
  const float4* r = (const float4*)(ea + (size_t)e * DD);
  const float4* vv = (const float4*)v;
  float s = 0.f;
#pragma unroll
  for (int k = 0; k < 16; ++k) {
    const float4 a = r[k], b = vv[k];
    s += a.x * b.x + a.y * b.y + a.z * b.z + a.w * b.w;
  }
  float l = asrc[row[e]] + adst[col[e]] + s;
  l = (l >= 0.f) ? l : NEG_SLOPE * l;   // leaky relu
  logit[e] = l;
  atomicMax(&menc[col[e]], encF(l));
}

// z = exp(logit - m[col]);  s[col] += z
__global__ void k_edge_exp(const float* __restrict__ logit,
                           const unsigned* __restrict__ menc,
                           const int* __restrict__ col,
                           float* __restrict__ z, float* __restrict__ s) {
  const int e = blockIdx.x * blockDim.x + threadIdx.x;
  if (e >= NE) return;
  const float m = decF(menc[col[e]]);
  const float zz = __expf(logit[e] - m);
  z[e] = zz;
  atomicAdd(&s[col[e]], zz);
}

// h[col] += alpha * xw[row]
__global__ void k_edge_scatter(const float* __restrict__ z,
                               const float* __restrict__ s,
                               const float* __restrict__ xw,
                               const int* __restrict__ row,
                               const int* __restrict__ col,
                               float* __restrict__ h) {
  const size_t idx = (size_t)blockIdx.x * blockDim.x + threadIdx.x;
  if (idx >= (size_t)NE * DD) return;
  const int e = (int)(idx >> 6);
  const int d = (int)(idx & 63);
  const float alpha = z[e] / (s[col[e]] + 1e-16f);
  atomicAdd(&h[(size_t)col[e] * DD + d], alpha * xw[(size_t)row[e] * DD + d]);
}

// BatchNorm stats: one block per channel
__global__ __launch_bounds__(256) void k_bn_stats(const float* __restrict__ h,
                                                  float* __restrict__ mu,
                                                  float* __restrict__ var) {
  __shared__ float rs[256], rq[256];
  const int d = blockIdx.x;
  const int tid = threadIdx.x;
  float s = 0.f, q = 0.f;
  for (int i = tid; i < NN; i += 256) {
    const float v = h[(size_t)i * DD + d];
    s += v; q += v * v;
  }
  rs[tid] = s; rq[tid] = q;
  __syncthreads();
  for (int o = 128; o > 0; o >>= 1) {
    if (tid < o) { rs[tid] += rs[tid + o]; rq[tid] += rq[tid + o]; }
    __syncthreads();
  }
  if (tid == 0) {
    const float m = rs[0] / (float)NN;
    mu[d] = m;
    var[d] = rq[0] / (float)NN - m * m;
  }
}

// x_out = relu((h - mu) * rsqrt(var+eps) * gamma + beta)
__global__ void k_bn_apply(const float* __restrict__ h,
                           const float* __restrict__ mu,
                           const float* __restrict__ var,
                           const float* __restrict__ gamma,
                           const float* __restrict__ beta,
                           float* __restrict__ xo) {
  const size_t idx = (size_t)blockIdx.x * blockDim.x + threadIdx.x;
  if (idx >= (size_t)NN * DD) return;
  const int d = (int)(idx & 63);
  float t = (h[idx] - mu[d]) * __frsqrt_rn(var[d] + EPS_BN) * gamma[d] + beta[d];
  xo[idx] = t > 0.f ? t : 0.f;
}

__global__ void k_pool_zero(float* __restrict__ pool, float* __restrict__ cnt) {
  const int i = blockIdx.x * blockDim.x + threadIdx.x;
  if (i < NG * DD) pool[i] = 0.f;
  if (i < NG) cnt[i] = 0.f;
}

__global__ void k_pool(const float* __restrict__ x, const int* __restrict__ batch,
                       float* __restrict__ pool, float* __restrict__ cnt) {
  const size_t idx = (size_t)blockIdx.x * blockDim.x + threadIdx.x;
  if (idx >= (size_t)NN * DD) return;
  const int i = (int)(idx >> 6);
  const int d = (int)(idx & 63);
  const int g = batch[i];
  atomicAdd(&pool[g * DD + d], x[idx]);
  if (d == 0) atomicAdd(&cnt[g], 1.f);
}

// readout MLP on (32 x 64) — tiny, single block
__global__ __launch_bounds__(256) void k_readout(
    const float* __restrict__ pool, const float* __restrict__ cnt,
    const float* __restrict__ w1, const float* __restrict__ b1,
    const float* __restrict__ w2, const float* __restrict__ b2,
    const float* __restrict__ w3, const float* __restrict__ b3,
    float* __restrict__ out) {
  __shared__ float sg[NG * DD], sh1[NG * DD], sh2[NG * 32];
  const int tid = threadIdx.x;
  for (int f = tid; f < NG * DD; f += 256) {
    const int i = f >> 6;
    sg[f] = pool[f] / fmaxf(cnt[i], 1.f);
  }
  __syncthreads();
  for (int f = tid; f < NG * DD; f += 256) {
    const int i = f >> 6, j = f & 63;
    float s = b1[j];
#pragma unroll 8
    for (int k = 0; k < DD; ++k) s += sg[i * DD + k] * w1[k * DD + j];
    sh1[f] = s > 0.f ? s : 0.f;
  }
  __syncthreads();
  for (int f = tid; f < NG * 32; f += 256) {
    const int i = f >> 5, j = f & 31;
    float s = b2[j];
#pragma unroll 8
    for (int k = 0; k < DD; ++k) s += sh1[i * DD + k] * w2[k * 32 + j];
    sh2[f] = s > 0.f ? s : 0.f;
  }
  __syncthreads();
  if (tid < NG) {
    float s = b3[0];
#pragma unroll 8
    for (int k = 0; k < 32; ++k) s += sh2[tid * 32 + k] * w3[k];
    out[tid] = s;
  }
}

// ---------------------------------------------------------------------------
extern "C" void kernel_launch(void* const* d_in, const int* in_sizes, int n_in,
                              void* d_out, int out_size, void* d_ws, size_t ws_size,
                              hipStream_t stream) {
  const float* x_in    = (const float*)d_in[0];
  const int*   eidx    = (const int*)d_in[1];
  const int*   row     = eidx;
  const int*   col     = eidx + NE;
  const float* e_in0   = (const float*)d_in[2];
  const int*   batch   = (const int*)d_in[3];
  const float* em_w1   = (const float*)d_in[4];
  const float* em_b1   = (const float*)d_in[5];
  const float* em_w2   = (const float*)d_in[6];
  const float* em_b2   = (const float*)d_in[7];
  const float* gat_w   = (const float*)d_in[8];
  const float* att_src = (const float*)d_in[9];
  const float* att_dst = (const float*)d_in[10];
  const float* edge_w  = (const float*)d_in[11];
  const float* att_edge= (const float*)d_in[12];
  const float* gat_bias= (const float*)d_in[13];
  const float* bn_gamma= (const float*)d_in[14];
  const float* bn_beta = (const float*)d_in[15];
  const float* mlp_w1  = (const float*)d_in[16];
  const float* mlp_b1  = (const float*)d_in[17];
  const float* mlp_w2  = (const float*)d_in[18];
  const float* mlp_b2  = (const float*)d_in[19];
  const float* mlp_w3  = (const float*)d_in[20];
  const float* mlp_b3  = (const float*)d_in[21];

  // workspace carve-out (floats)
  float* p = (float*)d_ws;
  float* ws_e    = p; p += (size_t)NE * DD;   // 51.2M
  float* ws_x    = p; p += (size_t)NN * DD;
  float* ws_xw   = p; p += (size_t)NN * DD;
  float* ws_h    = p; p += (size_t)NN * DD;
  float* ws_asrc = p; p += NN;
  float* ws_adst = p; p += NN;
  float* ws_v    = p; p += DD;
  unsigned* ws_menc = (unsigned*)p; p += NN;
  float* ws_s    = p; p += NN;
  float* ws_z    = p; p += NE;
  float* ws_lg   = p; p += NE;
  float* ws_mu   = p; p += DD;
  float* ws_var  = p; p += DD;
  float* ws_pool = p; p += NG * DD;
  float* ws_cnt  = p; p += NG;

  const int nd_blk = (NN * DD + 255) / 256;
  const int e_blk  = (NE + 255) / 256;
  const int n_blk  = (NN + 255) / 256;
  const int ed_blk = (int)(((size_t)NE * DD + 255) / 256);

  for (int l = 0; l < NL; ++l) {
    const float* x_l = (l == 0) ? x_in : ws_x;
    const float* e_l = (l == 0) ? e_in0 : ws_e;

    // 1) edge MLP (WMMA); grid-stride, weights LDS-resident; in-place safe l>0
    k_edge_mlp<<<2500, 256, 0, stream>>>(
        x_l, row, col, e_l,
        em_w1 + (size_t)l * 192 * DD, em_b1 + l * DD,
        em_w2 + (size_t)l * DD * DD,  em_b2 + l * DD, ws_e);

    // 2) GAT
    k_xw<<<784, 256, 0, stream>>>(x_l, gat_w + (size_t)l * DD * DD, ws_xw);
    k_node_attn<<<n_blk, 256, 0, stream>>>(ws_xw, att_src + l * DD, att_dst + l * DD,
                                           ws_asrc, ws_adst);
    k_edge_vec<<<1, 64, 0, stream>>>(edge_w + (size_t)l * DD * DD, att_edge + l * DD, ws_v);
    k_init_node<<<nd_blk, 256, 0, stream>>>(ws_s, ws_menc, ws_h, gat_bias + l * DD);
    k_edge_logits<<<e_blk, 256, 0, stream>>>(ws_e, ws_v, ws_asrc, ws_adst, row, col,
                                             ws_lg, ws_menc);
    k_edge_exp<<<e_blk, 256, 0, stream>>>(ws_lg, ws_menc, col, ws_z, ws_s);
    k_edge_scatter<<<ed_blk, 256, 0, stream>>>(ws_z, ws_s, ws_xw, row, col, ws_h);

    // 3) BatchNorm + ReLU -> ws_x
    k_bn_stats<<<DD, 256, 0, stream>>>(ws_h, ws_mu, ws_var);
    k_bn_apply<<<nd_blk, 256, 0, stream>>>(ws_h, ws_mu, ws_var,
                                           bn_gamma + l * DD, bn_beta + l * DD, ws_x);
  }

  // global mean pool + readout
  k_pool_zero<<<(NG * DD + 255) / 256, 256, 0, stream>>>(ws_pool, ws_cnt);
  k_pool<<<nd_blk, 256, 0, stream>>>(ws_x, batch, ws_pool, ws_cnt);
  k_readout<<<1, 256, 0, stream>>>(ws_pool, ws_cnt, mlp_w1, mlp_b1, mlp_w2, mlp_b2,
                                   mlp_w3, mlp_b3, (float*)d_out);
}
